// SAGEConvGATMLPModel_17712445128822
// MI455X (gfx1250) — matrix operand order repaired and checked by reading, not compiled, not run on hardware
//
#include <hip/hip_runtime.h>
#include <hip/hip_bf16.h>

// Model dims (fixed by reference)
#define IN_DIM  1024
#define FC_DIM  128
#define AC_DIM  20
#define HH      4
#define C1_DIM  10
#define C2_DIM  5
#define HC1     40   // HH*C1
#define HC2     20   // HH*C2
#define FCN_DIM 168  // FC + AC + HC2
#define KPAD    192  // FCN padded to mult of 32 for WMMA K
#define NPAD    176  // FCN padded to mult of 16 for WMMA N
#define OUT_DIM 3

typedef __attribute__((ext_vector_type(16))) _Float16 v16h;
typedef __attribute__((ext_vector_type(8)))  _Float16 v8h;
typedef __attribute__((ext_vector_type(8)))  float    v8f;

static __device__ __forceinline__ float leaky01(float v) { return v > 0.f ? v : 0.01f * v; }
static __device__ __forceinline__ float leaky02(float v) { return v > 0.f ? v : 0.2f  * v; }

__device__ void atomicMaxF(float* addr, float val) {
  int old = __float_as_int(*addr);
  while (__int_as_float(old) < val) {
    int prev = atomicCAS((int*)addr, old, __float_as_int(val));
    if (prev == old) break;
    old = prev;
  }
}

// ---------------- utility kernels ----------------
__global__ void fill_f32(float* p, float v, long n) {
  long i = (long)blockIdx.x * blockDim.x + threadIdx.x;
  long st = (long)gridDim.x * blockDim.x;
  for (; i < n; i += st) p[i] = v;
}

__global__ void cvt_feat(const float* __restrict__ f, _Float16* __restrict__ h,
                         long nvalid, long ntot) {
  long i = (long)blockIdx.x * blockDim.x + threadIdx.x;
  long st = (long)gridDim.x * blockDim.x;
  for (; i < ntot; i += st) h[i] = (i < nvalid) ? (_Float16)f[i] : (_Float16)0.f;
}

// WcatT[256][1024] = [Wl | Wr]^T in f16 (N-major so B fragments are contiguous)
__global__ void build_wcat(const float* __restrict__ Wl, const float* __restrict__ Wr,
                           _Float16* __restrict__ wT) {
  long i = (long)blockIdx.x * blockDim.x + threadIdx.x;
  long st = (long)gridDim.x * blockDim.x;
  for (; i < (long)256 * IN_DIM; i += st) {
    int n = (int)(i >> 10), k = (int)(i & 1023);
    float v = (n < FC_DIM) ? Wl[(size_t)k * FC_DIM + n] : Wr[(size_t)k * FC_DIM + (n - FC_DIM)];
    wT[i] = (_Float16)v;
  }
}

// W1T[NPAD][KPAD] = fc1_W^T, zero padded
__global__ void build_w1t(const float* __restrict__ W1, _Float16* __restrict__ wT) {
  long i = (long)blockIdx.x * blockDim.x + threadIdx.x;
  long st = (long)gridDim.x * blockDim.x;
  for (; i < (long)NPAD * KPAD; i += st) {
    int n = (int)(i / KPAD), k = (int)(i % KPAD);
    float v = (n < FCN_DIM && k < FCN_DIM) ? W1[(size_t)k * FCN_DIM + n] : 0.f;
    wT[i] = (_Float16)v;
  }
}

// ---------------- generic WMMA f16 GEMM ----------------
// C[M x ldc] = A[M x K] * B, with B given as BT[ntiles*16 x K] (row n = column n of B).
// One wave computes a 16 x 64 strip (4 accumulators sharing the A fragment).
__global__ __launch_bounds__(256) void gemm_wmma(
    const _Float16* __restrict__ A, const _Float16* __restrict__ BT,
    float* __restrict__ C, int mtiles, int ntiles, int K, int ldc) {
  int nSuper = (ntiles + 3) >> 2;
  int wave = blockIdx.x * (blockDim.x >> 5) + (threadIdx.x >> 5);
  int mIdx = wave / nSuper;
  int nsup = wave - mIdx * nSuper;
  if (mIdx >= mtiles) return;            // wave-uniform: EXEC stays all-ones for WMMA
  int lane = threadIdx.x & 31;
  int sel  = lane >> 4;                  // half-wave select
  int r16  = lane & 15;
  const _Float16* arow = A + (size_t)(mIdx * 16 + r16) * K;
  int nbase = nsup * 4;
  const _Float16* brow[4];
#pragma unroll
  for (int t = 0; t < 4; ++t) {
    int nt = nbase + t; if (nt >= ntiles) nt = 0;   // safe dup, result discarded
    brow[t] = BT + (size_t)(nt * 16 + r16) * K;
  }
  v8f acc[4] = {v8f{}, v8f{}, v8f{}, v8f{}};
  for (int k0 = 0; k0 < K; k0 += 32) {
    // A 16x32 f16 fragment: lane holds contiguous K = k0+sel*16+[0..15] of row r16
    v16h a = *(const v16h*)(arow + k0 + sel * 16);
    __builtin_prefetch(arow + k0 + 128, 0, 1);
#pragma unroll
    for (int t = 0; t < 4; ++t) {
      // B 32x16 fragment: lane holds contiguous K = k0+sel*16+[0..15] of column r16
      v16h b = *(const v16h*)(brow[t] + k0 + sel * 16);
      acc[t] = __builtin_amdgcn_wmma_f32_16x16x32_f16(
          false, a, false, b, (short)0, acc[t], false, false);
    }
  }
#pragma unroll
  for (int t = 0; t < 4; ++t) {
    int nt = nbase + t;
    if (nt < ntiles) {
      float* cb = C + (size_t)(mIdx * 16 + sel * 8) * ldc + nt * 16 + r16;
#pragma unroll
      for (int rr = 0; rr < 8; ++rr) cb[(size_t)rr * ldc] = acc[t][rr];
    }
  }
}

// ---------------- SAGE aggregation (post-transform) ----------------
__global__ void sage_agg(const int* __restrict__ edges, const float* __restrict__ PQ,
                         float* __restrict__ aggr, float* __restrict__ deg, int E) {
  int e = blockIdx.x;
  if (e >= E) return;
  int s = edges[e], d = edges[E + e];
  int j = threadIdx.x;                    // blockDim = 128
  atomicAdd(&aggr[(size_t)d * FC_DIM + j], PQ[(size_t)s * 256 + j]);
  if (j == 0) atomicAdd(&deg[d], 1.f);
}

__global__ void sage_x(const float* __restrict__ aggr, const float* __restrict__ deg,
                       const float* __restrict__ PQ, const float* __restrict__ bl,
                       const float* __restrict__ bg, const float* __restrict__ bb,
                       float* __restrict__ x, int N) {
  const float inv = 1.f / sqrtf(1.0f + 1e-5f);
  long i = (long)blockIdx.x * blockDim.x + threadIdx.x;
  long st = (long)gridDim.x * blockDim.x;
  for (; i < (long)N * FC_DIM; i += st) {
    int node = (int)(i >> 7), j = (int)(i & 127);
    float dg = fmaxf(deg[node], 1.f);
    float v = aggr[i] / dg + bl[j] + PQ[(size_t)node * 256 + FC_DIM + j];
    x[i] = v * (bg[j] * inv) + bb[j];
  }
}

// ---------------- GATv2 ----------------
__global__ void edge_deg_attr(const int* __restrict__ e2, const float* __restrict__ ef,
                              float* __restrict__ deg2, float* __restrict__ sattr, int E) {
  long e = (long)blockIdx.x * blockDim.x + threadIdx.x;
  long st = (long)gridDim.x * blockDim.x;
  for (; e < E; e += st) {
    int d = e2[E + e];
    atomicAdd(&deg2[d], 1.f);
    atomicAdd(&sattr[d], ef[e]);
  }
}

// xl = xin@Wl + bl ; xr = xin@Wr + br   (tiny per-node linears)
__global__ void gat_lin(const float* __restrict__ xin, int DI, int HC,
                        const float* __restrict__ Wl, const float* __restrict__ bl,
                        const float* __restrict__ Wr, const float* __restrict__ br,
                        float* __restrict__ xl, float* __restrict__ xr, int N) {
  __shared__ float row[48];
  int node = blockIdx.x;
  if (node >= N) return;
  if (threadIdx.x < DI) row[threadIdx.x] = xin[(size_t)node * DI + threadIdx.x];
  __syncthreads();
  for (int t = threadIdx.x; t < 2 * HC; t += blockDim.x) {
    bool left = t < HC;
    int c = left ? t : t - HC;
    const float* W = left ? Wl : Wr;
    float acc = left ? bl[c] : br[c];
    for (int j = 0; j < DI; ++j) acc += row[j] * W[j * HC + c];
    (left ? xl : xr)[(size_t)node * HC + c] = acc;
  }
}

__global__ void gat_logit(const int* __restrict__ e2, const float* __restrict__ ef,
                          const float* __restrict__ deg2, const float* __restrict__ sattr,
                          const float* __restrict__ xl, const float* __restrict__ xr,
                          const float* __restrict__ We, const float* __restrict__ att,
                          float* __restrict__ elog, float* __restrict__ mx,
                          int E, int N, int H, int C) {
  long EP = (long)E + N;
  long e = (long)blockIdx.x * blockDim.x + threadIdx.x;
  long st = (long)gridDim.x * blockDim.x;
  int HC = H * C;
  for (; e < EP; e += st) {
    int s, d; float a;
    if (e < E) { s = e2[e]; d = e2[E + e]; a = ef[e]; }
    else       { s = d = (int)(e - E); a = sattr[s] / fmaxf(deg2[s], 1.f); }
    const float* pl = xl + (size_t)s * HC;
    const float* pr = xr + (size_t)d * HC;
    for (int h = 0; h < H; ++h) {
      float lg = 0.f;
      for (int c = 0; c < C; ++c) {
        int q = h * C + c;
        float m = leaky02(pl[q] + pr[q] + a * We[q]);
        lg += m * att[q];
      }
      elog[e * H + h] = lg;
      atomicMaxF(&mx[(size_t)d * H + h], lg);
    }
  }
}

__global__ void gat_exp(const int* __restrict__ e2, float* __restrict__ elog,
                        const float* __restrict__ mx, float* __restrict__ den,
                        int E, int N, int H) {
  long tot = ((long)E + N) * H;
  long i = (long)blockIdx.x * blockDim.x + threadIdx.x;
  long st = (long)gridDim.x * blockDim.x;
  for (; i < tot; i += st) {
    long e = i / H; int h = (int)(i - e * H);
    int d = (e < E) ? e2[E + e] : (int)(e - E);
    float ex = expf(elog[i] - mx[(size_t)d * H + h]);
    elog[i] = ex;
    atomicAdd(&den[(size_t)d * H + h], ex);
  }
}

__global__ void gat_scatter(const int* __restrict__ e2, const float* __restrict__ elog,
                            const float* __restrict__ den, const float* __restrict__ xl,
                            float* __restrict__ yacc, int E, int N, int H, int C) {
  int HC = H * C;
  long tot = ((long)E + N) * HC;
  long i = (long)blockIdx.x * blockDim.x + threadIdx.x;
  long st = (long)gridDim.x * blockDim.x;
  for (; i < tot; i += st) {
    long e = i / HC; int q = (int)(i - e * HC); int h = q / C;
    int s, d;
    if (e < E) { s = e2[e]; d = e2[E + e]; }
    else       { s = d = (int)(e - E); }
    float alpha = elog[e * H + h] / (den[(size_t)d * H + h] + 1e-16f);
    atomicAdd(&yacc[(size_t)d * HC + q], xl[(size_t)s * HC + q] * alpha);
  }
}

__global__ void gat_finish1(float* __restrict__ y, const float* __restrict__ b, int N, int HC) {
  long i = (long)blockIdx.x * blockDim.x + threadIdx.x;
  long st = (long)gridDim.x * blockDim.x;
  for (; i < (long)N * HC; i += st) {
    int q = (int)(i % HC);
    y[i] = leaky01(y[i] + b[q]);
  }
}

__global__ void gat_finish2(float* __restrict__ y, const float* __restrict__ b,
                            const float* __restrict__ bg, const float* __restrict__ bb,
                            int N, int HC) {
  const float inv = 1.f / sqrtf(1.0f + 1e-5f);
  long i = (long)blockIdx.x * blockDim.x + threadIdx.x;
  long st = (long)gridDim.x * blockDim.x;
  for (; i < (long)N * HC; i += st) {
    int q = (int)(i % HC);
    y[i] = (y[i] + b[q]) * (bg[q] * inv) + bb[q];
  }
}

// ---------------- MLP head ----------------
__global__ void build_zh(const float* __restrict__ x, const float* __restrict__ y2,
                         const float* __restrict__ af, _Float16* __restrict__ zH,
                         int N, long Nr) {
  long tot = Nr * KPAD;
  long i = (long)blockIdx.x * blockDim.x + threadIdx.x;
  long st = (long)gridDim.x * blockDim.x;
  for (; i < tot; i += st) {
    long node = i / KPAD; int j = (int)(i - node * KPAD);
    float t = 0.f;
    if (node < N) {
      if (j < FC_DIM)            t = leaky01(x[node * FC_DIM + j]);
      else if (j < FC_DIM + HC2) t = leaky01(y2[node * HC2 + (j - FC_DIM)]);
      else if (j < FCN_DIM)      t = af[node * AC_DIM + (j - FC_DIM - HC2)];
    }
    zH[i] = (_Float16)t;
  }
}

__global__ void head_final(const float* __restrict__ T1, const float* __restrict__ b1,
                           const float* __restrict__ bg, const float* __restrict__ bb,
                           const float* __restrict__ W2, const float* __restrict__ b2,
                           float* __restrict__ out, int N) {
  const float inv = 1.f / sqrtf(1.0f + 1e-5f);
  long i = (long)blockIdx.x * blockDim.x + threadIdx.x;
  long st = (long)gridDim.x * blockDim.x;
  for (; i < N; i += st) {
    const float* trow = T1 + i * (long)NPAD;
    float a0 = b2[0], a1 = b2[1], a2 = b2[2];
    for (int k = 0; k < FCN_DIM; ++k) {
      float rv = (trow[k] + b1[k]) * (bg[k] * inv) + bb[k];
      rv = fmaxf(rv, 0.f);
      a0 += rv * W2[k * OUT_DIM + 0];
      a1 += rv * W2[k * OUT_DIM + 1];
      a2 += rv * W2[k * OUT_DIM + 2];
    }
    out[i * OUT_DIM + 0] = a0;
    out[i * OUT_DIM + 1] = a1;
    out[i * OUT_DIM + 2] = a2;
  }
}

// ---------------- launch ----------------
static inline unsigned gsz(long n, int t) {
  long b = (n + t - 1) / t;
  if (b < 1) b = 1;
  if (b > 262144) b = 262144;
  return (unsigned)b;
}

extern "C" void kernel_launch(void* const* d_in, const int* in_sizes, int n_in,
                              void* d_out, int out_size, void* d_ws, size_t ws_size,
                              hipStream_t stream) {
  const float* features = (const float*)d_in[0];
  const int*   edges    = (const int*)d_in[1];
  const int*   edges2   = (const int*)d_in[2];
  const float* ef       = (const float*)d_in[3];
  const float* af       = (const float*)d_in[4];
  const float* sage_Wl  = (const float*)d_in[5];
  const float* sage_bl  = (const float*)d_in[6];
  const float* sage_Wr  = (const float*)d_in[7];
  const float* g1_Wl = (const float*)d_in[8],  *g1_bl = (const float*)d_in[9];
  const float* g1_Wr = (const float*)d_in[10], *g1_br = (const float*)d_in[11];
  const float* g1_We = (const float*)d_in[12], *g1_att = (const float*)d_in[13];
  const float* g1_b  = (const float*)d_in[14];
  const float* g2_Wl = (const float*)d_in[15], *g2_bl = (const float*)d_in[16];
  const float* g2_Wr = (const float*)d_in[17], *g2_br = (const float*)d_in[18];
  const float* g2_We = (const float*)d_in[19], *g2_att = (const float*)d_in[20];
  const float* g2_b  = (const float*)d_in[21];
  const float* fc1_W = (const float*)d_in[22], *fc1_b = (const float*)d_in[23];
  const float* fc2_W = (const float*)d_in[24], *fc2_b = (const float*)d_in[25];
  const float* bnx_g = (const float*)d_in[26], *bnx_b = (const float*)d_in[27];
  const float* bny_g = (const float*)d_in[28], *bny_b = (const float*)d_in[29];
  const float* bnf_g = (const float*)d_in[30], *bnf_b = (const float*)d_in[31];
  float* out = (float*)d_out;

  const int N  = in_sizes[4] / AC_DIM;   // 50000
  const int E  = in_sizes[3];            // 300000
  const long Nr = ((long)N + 15) & ~15L; // rows padded to 16 for WMMA
  const int mtiles = (int)(Nr / 16);

  // -------- workspace layout (bump allocator; region A recycled) --------
  char* base = (char*)d_ws;
  size_t off = 0;
  auto bump = [&](size_t bytes) { size_t o = off; off += (bytes + 255) & ~(size_t)255; return o; };
  size_t oFeatH = bump(Nr * IN_DIM * 2);        // f16 features (recycled after SAGE gemm)
  size_t oWcat  = bump((size_t)256 * IN_DIM * 2);
  size_t oPQ    = bump(Nr * 256 * 4);           // [P | Q] f32
  size_t oAggr  = bump((size_t)N * FC_DIM * 4);
  size_t oDeg   = bump((size_t)N * 4);
  size_t oX     = bump((size_t)N * FC_DIM * 4);
  // region A: inside the f16 feature span, valid once the SAGE gemm is done
  size_t ra = oFeatH;
  auto bumpA = [&](size_t bytes) { size_t o = ra; ra += (bytes + 255) & ~(size_t)255; return o; };
  size_t oXl1  = bumpA((size_t)N * HC1 * 4);
  size_t oXr1  = bumpA((size_t)N * HC1 * 4);
  size_t oElog = bumpA(((size_t)E + N) * HH * 4);
  size_t oMx   = bumpA((size_t)N * HH * 4);
  size_t oDen  = bumpA((size_t)N * HH * 4);
  size_t oY1   = bumpA((size_t)N * HC1 * 4);
  size_t oXl2  = bumpA((size_t)N * HC2 * 4);
  size_t oXr2  = bumpA((size_t)N * HC2 * 4);
  size_t oY2   = bumpA((size_t)N * HC2 * 4);
  size_t oDeg2 = bumpA((size_t)N * 4);
  size_t oSat  = bumpA((size_t)N * 4);
  size_t oZH   = bumpA(Nr * KPAD * 2);
  size_t oW1T  = bumpA((size_t)NPAD * KPAD * 2);
  size_t oT1   = bumpA(Nr * NPAD * 4);

  _Float16* featH = (_Float16*)(base + oFeatH);
  _Float16* wcatT = (_Float16*)(base + oWcat);
  float* PQ   = (float*)(base + oPQ);
  float* aggr = (float*)(base + oAggr);
  float* deg  = (float*)(base + oDeg);
  float* x    = (float*)(base + oX);
  float* xl1 = (float*)(base + oXl1), *xr1 = (float*)(base + oXr1);
  float* elog = (float*)(base + oElog);
  float* mx = (float*)(base + oMx), *den = (float*)(base + oDen);
  float* y1 = (float*)(base + oY1);
  float* xl2 = (float*)(base + oXl2), *xr2 = (float*)(base + oXr2);
  float* y2 = (float*)(base + oY2);
  float* deg2 = (float*)(base + oDeg2), *sattr = (float*)(base + oSat);
  _Float16* zH  = (_Float16*)(base + oZH);
  _Float16* w1T = (_Float16*)(base + oW1T);
  float* T1 = (float*)(base + oT1);

  // -------- SAGE: transform -> aggregate --------
  cvt_feat<<<gsz(Nr * IN_DIM, 256), 256, 0, stream>>>(features, featH,
                                                      (long)N * IN_DIM, Nr * IN_DIM);
  build_wcat<<<gsz((long)256 * IN_DIM, 256), 256, 0, stream>>>(sage_Wl, sage_Wr, wcatT);
  {
    long waves = (long)mtiles * 4;            // 16 n-tiles -> 4 supertiles
    gemm_wmma<<<gsz(waves * 32, 256), 256, 0, stream>>>(featH, wcatT, PQ,
                                                        mtiles, 16, IN_DIM, 256);
  }
  fill_f32<<<gsz((long)N * FC_DIM, 256), 256, 0, stream>>>(aggr, 0.f, (long)N * FC_DIM);
  fill_f32<<<gsz(N, 256), 256, 0, stream>>>(deg, 0.f, N);
  sage_agg<<<E, FC_DIM, 0, stream>>>(edges, PQ, aggr, deg, E);
  sage_x<<<gsz((long)N * FC_DIM, 256), 256, 0, stream>>>(aggr, deg, PQ, sage_bl,
                                                         bnx_g, bnx_b, x, N);

  // -------- self-loop mean attrs --------
  fill_f32<<<gsz(N, 256), 256, 0, stream>>>(deg2, 0.f, N);
  fill_f32<<<gsz(N, 256), 256, 0, stream>>>(sattr, 0.f, N);
  edge_deg_attr<<<gsz(E, 256), 256, 0, stream>>>(edges2, ef, deg2, sattr, E);

  const long EP = (long)E + N;

  // -------- GATv2 layer 1 (20 -> 4x10) --------
  gat_lin<<<N, 128, 0, stream>>>(af, AC_DIM, HC1, g1_Wl, g1_bl, g1_Wr, g1_br, xl1, xr1, N);
  fill_f32<<<gsz((long)N * HH, 256), 256, 0, stream>>>(mx, -3.0e38f, (long)N * HH);
  fill_f32<<<gsz((long)N * HH, 256), 256, 0, stream>>>(den, 0.f, (long)N * HH);
  gat_logit<<<gsz(EP, 256), 256, 0, stream>>>(edges2, ef, deg2, sattr, xl1, xr1,
                                              g1_We, g1_att, elog, mx, E, N, HH, C1_DIM);
  gat_exp<<<gsz(EP * HH, 256), 256, 0, stream>>>(edges2, elog, mx, den, E, N, HH);
  fill_f32<<<gsz((long)N * HC1, 256), 256, 0, stream>>>(y1, 0.f, (long)N * HC1);
  gat_scatter<<<gsz(EP * HC1, 256), 256, 0, stream>>>(edges2, elog, den, xl1, y1,
                                                      E, N, HH, C1_DIM);
  gat_finish1<<<gsz((long)N * HC1, 256), 256, 0, stream>>>(y1, g1_b, N, HC1);

  // -------- GATv2 layer 2 (4x10 -> 4x5) --------
  gat_lin<<<N, 128, 0, stream>>>(y1, HC1, HC2, g2_Wl, g2_bl, g2_Wr, g2_br, xl2, xr2, N);
  fill_f32<<<gsz((long)N * HH, 256), 256, 0, stream>>>(mx, -3.0e38f, (long)N * HH);
  fill_f32<<<gsz((long)N * HH, 256), 256, 0, stream>>>(den, 0.f, (long)N * HH);
  gat_logit<<<gsz(EP, 256), 256, 0, stream>>>(edges2, ef, deg2, sattr, xl2, xr2,
                                              g2_We, g2_att, elog, mx, E, N, HH, C2_DIM);
  gat_exp<<<gsz(EP * HH, 256), 256, 0, stream>>>(edges2, elog, mx, den, E, N, HH);
  fill_f32<<<gsz((long)N * HC2, 256), 256, 0, stream>>>(y2, 0.f, (long)N * HC2);
  gat_scatter<<<gsz(EP * HC2, 256), 256, 0, stream>>>(edges2, elog, den, xl2, y2,
                                                      E, N, HH, C2_DIM);
  gat_finish2<<<gsz((long)N * HC2, 256), 256, 0, stream>>>(y2, g2_b, bny_g, bny_b, N, HC2);

  // -------- MLP head: fc1 via WMMA, then bn/relu/fc2 --------
  build_zh<<<gsz(Nr * KPAD, 256), 256, 0, stream>>>(x, y2, af, zH, N, Nr);
  build_w1t<<<gsz((long)NPAD * KPAD, 256), 256, 0, stream>>>(fc1_W, w1T);
  {
    int ntiles = NPAD / 16;                   // 11
    int nSuper = (ntiles + 3) >> 2;           // 3
    long waves = (long)mtiles * nSuper;
    gemm_wmma<<<gsz(waves * 32, 256), 256, 0, stream>>>(zH, w1T, T1,
                                                        mtiles, ntiles, KPAD, NPAD);
  }
  head_final<<<gsz(N, 256), 256, 0, stream>>>(T1, fc1_b, bnf_g, bnf_b, fc2_W, fc2_b, out, N);
}
